// ShipmentGraphSAGE_77498389889595
// MI455X (gfx1250) — matrix operand order repaired and checked by reading, not compiled, not run on hardware
//
#include <hip/hip_runtime.h>

typedef __attribute__((ext_vector_type(16))) _Float16 v16h;
typedef __attribute__((ext_vector_type(8)))  float    v8f;

union HFrag { v16h v; _Float16 h[16]; float4 q[2]; };
union FAcc  { v8f  v; float    f[8];  };

// ---------------------------------------------------------------------------
// Zero a float buffer (grid-stride).
// ---------------------------------------------------------------------------
__global__ void zero_f32(float* __restrict__ p, long long n) {
  long long i = (long long)blockIdx.x * blockDim.x + threadIdx.x;
  long long stride = (long long)gridDim.x * blockDim.x;
  for (; i < n; i += stride) p[i] = 0.0f;
}

// ---------------------------------------------------------------------------
// Edge scatter: agg[dst] += feat[src]  (+ optional degree count).
// 16 lanes per edge, each lane moves one float4 (4 channels).
// ---------------------------------------------------------------------------
__global__ void scatter_accum(const float* __restrict__ feat,
                              const int*   __restrict__ src,
                              const int*   __restrict__ dst,
                              float* __restrict__ agg,
                              float* __restrict__ deg,   // may be null
                              long long nEdges) {
  long long idx = (long long)blockIdx.x * blockDim.x + threadIdx.x;
  long long e = idx >> 4;
  int lane16 = (int)(idx & 15);
  if (e >= nEdges) return;
  int s = src[e];
  int d = dst[e];
  const float4 v = *(const float4*)(feat + (size_t)s * 64 + lane16 * 4);
  float* base = agg + (size_t)d * 64 + lane16 * 4;
  atomicAdd(base + 0, v.x);
  atomicAdd(base + 1, v.y);
  atomicAdd(base + 2, v.z);
  atomicAdd(base + 3, v.w);
  if (deg != nullptr && lane16 == 0) atomicAdd(deg + d, 1.0f);
}

// ---------------------------------------------------------------------------
// A fragment (16x32 f16) for V_WMMA_F32_16X16X32_F16.
// Lane L: m = L%16, kh = L/16. Element j: k = kbase + (j<8 ? 8*kh + j
//                                                      : 16 + 8*kh + (j-8)).
// ---------------------------------------------------------------------------
__device__ inline v16h make_afrag(const float* __restrict__ rowPtr,
                                  int kbase, int kh, float scale) {
  const float* p = rowPtr + kbase + 8 * kh;
  float4 q0 = *(const float4*)(p);
  float4 q1 = *(const float4*)(p + 4);
  float4 q2 = *(const float4*)(p + 16);
  float4 q3 = *(const float4*)(p + 20);
  HFrag f;
  f.h[0]  = (_Float16)(q0.x * scale); f.h[1]  = (_Float16)(q0.y * scale);
  f.h[2]  = (_Float16)(q0.z * scale); f.h[3]  = (_Float16)(q0.w * scale);
  f.h[4]  = (_Float16)(q1.x * scale); f.h[5]  = (_Float16)(q1.y * scale);
  f.h[6]  = (_Float16)(q1.z * scale); f.h[7]  = (_Float16)(q1.w * scale);
  f.h[8]  = (_Float16)(q2.x * scale); f.h[9]  = (_Float16)(q2.y * scale);
  f.h[10] = (_Float16)(q2.z * scale); f.h[11] = (_Float16)(q2.w * scale);
  f.h[12] = (_Float16)(q3.x * scale); f.h[13] = (_Float16)(q3.y * scale);
  f.h[14] = (_Float16)(q3.z * scale); f.h[15] = (_Float16)(q3.w * scale);
  return f.v;
}

// ---------------------------------------------------------------------------
// Swizzled-LDS B fragment: each lane's 16 halves are 32 contiguous bytes,
// fetched as two ds_load_b128.
// Swizzle: element W[k][n] -> frag(c=k/32, ct=n/16), lane L=16*((k>>3)&1)
//          + (n&15), slot j=(k&7)+8*((k>>4)&1), addr = ((c*4+ct)*32 + L)*16 + j.
// ---------------------------------------------------------------------------
__device__ inline v16h make_bfrag(const _Float16* __restrict__ sW,
                                  int c, int ct, int lane) {
  const _Float16* p = sW + (size_t)(((c * 4 + ct) * 32 + lane) * 16);
  HFrag f;
  f.q[0] = *(const float4*)(p);
  f.q[1] = *(const float4*)(p + 8);
  return f.v;
}

// ---------------------------------------------------------------------------
// SAGE layer: out = relu( (agg/deg) @ Wl + b + xin @ Wr )
// HEAD=true additionally fuses logits = h @ w_head + b_head (out is [N]).
// 256 threads = 8 waves; each wave computes one 16-row tile.
// ---------------------------------------------------------------------------
template <bool HEAD>
__global__ void __launch_bounds__(256)
sage_layer_wmma(const float* __restrict__ agg,
                const float* __restrict__ deg,
                const float* __restrict__ xin,
                const float* __restrict__ wl,
                const float* __restrict__ wr,
                const float* __restrict__ bias,
                const float* __restrict__ whead,
                const float* __restrict__ bhead,
                float* __restrict__ outp,
                int nTiles) {
  __shared__ _Float16 sWl[64 * 64];   // swizzled fragment layout
  __shared__ _Float16 sWr[64 * 64];
  __shared__ float sB[64];
  __shared__ float sWh[64];

  // Stage weights into LDS, converted to f16 and pre-swizzled into the
  // WMMA B-fragment layout.
  for (int i = threadIdx.x; i < 64 * 64; i += blockDim.x) {
    const int k  = i >> 6;
    const int n  = i & 63;
    const int c  = k >> 5;            // K-chunk of 32
    const int hh = (k >> 3) & 1;      // K half within chunk
    const int j  = (k & 7) + 8 * ((k >> 4) & 1);  // slot within lane
    const int ct = n >> 4;            // col tile
    const int L  = hh * 16 + (n & 15);
    const int a  = ((c * 4 + ct) * 32 + L) * 16 + j;
    sWl[a] = (_Float16)wl[i];
    sWr[a] = (_Float16)wr[i];
  }
  for (int i = threadIdx.x; i < 64; i += blockDim.x) {
    sB[i]  = bias[i];
    sWh[i] = HEAD ? whead[i] : 0.0f;
  }
  __syncthreads();

  const int wave = threadIdx.x >> 5;
  const int lane = threadIdx.x & 31;
  const int tile = blockIdx.x * (blockDim.x >> 5) + wave;
  if (tile >= nTiles) return;   // wave-uniform: EXEC stays all-ones for WMMA

  const int row0   = tile * 16;
  const int mLocal = lane & 15;
  const int kh     = lane >> 4;
  const int m      = row0 + mLocal;
  const size_t rowOff = (size_t)m * 64;

  const float invd = 1.0f / fmaxf(deg[m], 1.0f);

  // A fragments: 2 K-chunks of scaled aggregation, 2 K-chunks of xin.
  const v16h aA0 = make_afrag(agg + rowOff, 0,  kh, invd);
  const v16h aA1 = make_afrag(agg + rowOff, 32, kh, invd);
  const v16h aX0 = make_afrag(xin + rowOff, 0,  kh, 1.0f);
  const v16h aX1 = make_afrag(xin + rowOff, 32, kh, 1.0f);

  float pacc[8];
  if (HEAD) {
#pragma unroll
    for (int r = 0; r < 8; ++r) pacc[r] = 0.0f;
  }

#pragma unroll
  for (int ct = 0; ct < 4; ++ct) {
    const int n = ct * 16 + mLocal;
    FAcc acc;
    const float bv = sB[n];
#pragma unroll
    for (int r = 0; r < 8; ++r) acc.f[r] = bv;   // bias folded into C

    v16h b;
    b = make_bfrag(sWl, 0, ct, lane);
    acc.v = __builtin_amdgcn_wmma_f32_16x16x32_f16(false, aA0, false, b,
                                                   (short)0, acc.v, false, false);
    b = make_bfrag(sWl, 1, ct, lane);
    acc.v = __builtin_amdgcn_wmma_f32_16x16x32_f16(false, aA1, false, b,
                                                   (short)0, acc.v, false, false);
    b = make_bfrag(sWr, 0, ct, lane);
    acc.v = __builtin_amdgcn_wmma_f32_16x16x32_f16(false, aX0, false, b,
                                                   (short)0, acc.v, false, false);
    b = make_bfrag(sWr, 1, ct, lane);
    acc.v = __builtin_amdgcn_wmma_f32_16x16x32_f16(false, aX1, false, b,
                                                   (short)0, acc.v, false, false);

#pragma unroll
    for (int r = 0; r < 8; ++r) acc.f[r] = fmaxf(acc.f[r], 0.0f);  // ReLU

    if (HEAD) {
      const float wv = sWh[n];
#pragma unroll
      for (int r = 0; r < 8; ++r) pacc[r] += acc.f[r] * wv;
    } else {
      // C/D layout: m = row0 + r + 8*kh, n = ct*16 + lane%16
#pragma unroll
      for (int r = 0; r < 8; ++r) {
        const int mm = row0 + r + 8 * kh;
        outp[(size_t)mm * 64 + n] = acc.f[r];
      }
    }
  }

  if (HEAD) {
    // Reduce over the 16 lanes of each half-wave (covers all 64 columns).
#pragma unroll
    for (int mask = 1; mask < 16; mask <<= 1) {
#pragma unroll
      for (int r = 0; r < 8; ++r) pacc[r] += __shfl_xor(pacc[r], mask, 32);
    }
    if (mLocal == 0) {
      const float bh = bhead[0];
#pragma unroll
      for (int r = 0; r < 8; ++r) outp[row0 + r + 8 * kh] = pacc[r] + bh;
    }
  }
}

// ---------------------------------------------------------------------------
// Launch: zero -> scatter(x)+deg -> layer1(WMMA) -> zero -> scatter(h1)
//         -> layer2(WMMA)+head
// ---------------------------------------------------------------------------
extern "C" void kernel_launch(void* const* d_in, const int* in_sizes, int n_in,
                              void* d_out, int out_size, void* d_ws, size_t ws_size,
                              hipStream_t stream) {
  const float* x    = (const float*)d_in[0];
  const int*   ei   = (const int*)d_in[1];
  const float* w1l  = (const float*)d_in[2];
  const float* b1   = (const float*)d_in[3];
  const float* w1r  = (const float*)d_in[4];
  const float* w2l  = (const float*)d_in[5];
  const float* b2   = (const float*)d_in[6];
  const float* w2r  = (const float*)d_in[7];
  const float* wh   = (const float*)d_in[8];
  const float* bh   = (const float*)d_in[9];
  float* out = (float*)d_out;

  const long long N = (long long)in_sizes[0] / 64;
  const long long E = (long long)in_sizes[1] / 2;
  const int* src = ei;
  const int* dst = ei + E;

  float* deg = (float*)d_ws;                 // [N]
  float* agg = deg + N;                      // [N,64]
  float* h1  = agg + N * 64;                 // [N,64]

  // Zero deg + agg (contiguous).
  {
    long long n = N + N * 64;
    int blocks = (int)((n + 255) / 256);
    if (blocks > 65535) blocks = 65535;
    zero_f32<<<blocks, 256, 0, stream>>>(deg, n);
  }

  // Layer 1 aggregation (+ degrees).
  {
    long long threads = E * 16;
    int blocks = (int)((threads + 255) / 256);
    scatter_accum<<<blocks, 256, 0, stream>>>(x, src, dst, agg, deg, E);
  }

  const int nTiles = (int)(N / 16);
  const int layerBlocks = (nTiles + 7) / 8;

  sage_layer_wmma<false><<<layerBlocks, 256, 0, stream>>>(
      agg, deg, x, w1l, w1r, b1, nullptr, nullptr, h1, nTiles);

  // Re-zero agg for layer 2.
  {
    long long n = N * 64;
    int blocks = (int)((n + 255) / 256);
    if (blocks > 65535) blocks = 65535;
    zero_f32<<<blocks, 256, 0, stream>>>(agg, n);
  }

  // Layer 2 aggregation (degrees already computed).
  {
    long long threads = E * 16;
    int blocks = (int)((threads + 255) / 256);
    scatter_accum<<<blocks, 256, 0, stream>>>(h1, src, dst, agg, nullptr, E);
  }

  sage_layer_wmma<true><<<layerBlocks, 256, 0, stream>>>(
      agg, deg, h1, w2l, w2r, b2, wh, bh, out, nTiles);
}
